// MultiHeadAttentionEinsum_15015205667434
// MI455X (gfx1250) — compile-verified, matrix operand order checked by reading, not run on hardware
//
#include <hip/hip_runtime.h>

// ---------------------------------------------------------------------------
// MHA forward for B=2,S=2048,D=1024,H=16,HD=64 on gfx1250 (CDNA5, wave32).
// All matmuls on v_wmma_f32_16x16x32_bf16 (f32 accumulate). Tile staging uses
// GLOBAL_LOAD_ASYNC_TO_LDS_B128 when the toolchain exposes the builtin.
// Workspace layout (needs 48 MB):
//   [0,8)MB  xbf          [8,10) WqT  [10,12) WkT  [12,14) WvT  [14,16) Wo(bf)
//   [16,24)  Q [B,H,S,HD] [24,32) K   [32,40) V    [40,48) ctx [B*S, D] bf16
// ---------------------------------------------------------------------------

typedef __attribute__((ext_vector_type(16))) __bf16 v16bf;
typedef __attribute__((ext_vector_type(8)))  __bf16 v8bf;
typedef __attribute__((ext_vector_type(8)))  float  v8f;

constexpr int Bk = 2, Sk = 2048, Dk = 1024, Hk = 16, HDk = 64;

#if defined(__AMDGCN__) && __has_builtin(__builtin_amdgcn_global_load_async_to_lds_b128)
#define USE_ASYNC 1
#else
#define USE_ASYNC 0
#endif

#if USE_ASYNC
// builtin signature (from clang diagnostic): param0 = int4* in AS(1) (global);
// param1 = LDS-side pointer; then imm offset, imm cpol.
typedef int v4i_g __attribute__((vector_size(16)));
typedef __attribute__((address_space(1))) v4i_g GVec;
typedef __attribute__((address_space(3))) v4i_g LVec;
// per-lane 16B global -> LDS async copy (ASYNCcnt-tracked, no VGPR data)
__device__ __forceinline__ void cp_async16(__bf16* lds, const __bf16* g) {
  __builtin_amdgcn_global_load_async_to_lds_b128(
      (GVec*)(unsigned long long)(uintptr_t)g,
      (LVec*)(unsigned int)(uintptr_t)lds, 0, 0);
}
__device__ __forceinline__ void wait_async() {
#if __has_builtin(__builtin_amdgcn_s_wait_asynccnt)
  __builtin_amdgcn_s_wait_asynccnt(0);
#else
  asm volatile("s_wait_asynccnt 0x0" ::: "memory");
#endif
}
#endif

// copy one 16B chunk global->LDS (async when available)
__device__ __forceinline__ void stage16(__bf16* lds, const __bf16* g) {
#if USE_ASYNC
  cp_async16(lds, g);
#else
  *(v8bf*)lds = *(const v8bf*)g;
#endif
}
__device__ __forceinline__ void stage_fence() {
#if USE_ASYNC
  wait_async();
#endif
}

__device__ __forceinline__ v16bf cat8(v8bf a, v8bf b) {
  return __builtin_shufflevector(a, b, 0,1,2,3,4,5,6,7,8,9,10,11,12,13,14,15);
}
__device__ __forceinline__ v8f wmma_bf16(v16bf a, v16bf b, v8f c) {
  return __builtin_amdgcn_wmma_f32_16x16x32_bf16(false, a, false, b,
                                                 (short)0, c, false, false);
}
__device__ __forceinline__ float redmax16(float x) {
  x = fmaxf(x, __shfl_xor(x, 1, 16));
  x = fmaxf(x, __shfl_xor(x, 2, 16));
  x = fmaxf(x, __shfl_xor(x, 4, 16));
  x = fmaxf(x, __shfl_xor(x, 8, 16));
  return x;
}
__device__ __forceinline__ float redsum16(float x) {
  x += __shfl_xor(x, 1, 16);
  x += __shfl_xor(x, 2, 16);
  x += __shfl_xor(x, 4, 16);
  x += __shfl_xor(x, 8, 16);
  return x;
}
__device__ __forceinline__ float fexp2(float x) {
#if defined(__AMDGCN__) && __has_builtin(__builtin_amdgcn_exp2f)
  return __builtin_amdgcn_exp2f(x);   // bare v_exp_f32
#else
  return exp2f(x);
#endif
}

// --------------------------- prep kernels ----------------------------------

__global__ void k_cvt_f32_bf16(const float* __restrict__ in,
                               __bf16* __restrict__ out, int n) {
  int i = (blockIdx.x * blockDim.x + threadIdx.x) * 4;
  if (i + 3 < n) {
    float4 v = *(const float4*)&in[i];
    out[i + 0] = (__bf16)v.x;
    out[i + 1] = (__bf16)v.y;
    out[i + 2] = (__bf16)v.z;
    out[i + 3] = (__bf16)v.w;
  }
}

// WT[i][d] = (bf16)W[d][i], square n x n, block (32,8), grid (n/32, n/32)
__global__ void k_tr_cvt(const float* __restrict__ W,
                         __bf16* __restrict__ WT, int n) {
  __shared__ float tile[32][33];
  int x  = blockIdx.x * 32 + threadIdx.x;  // col of W (= row of WT)
  int y0 = blockIdx.y * 32;                // row block of W
  for (int j = threadIdx.y; j < 32; j += 8)
    tile[j][threadIdx.x] = W[(size_t)(y0 + j) * n + x];
  __syncthreads();
  int yo0 = blockIdx.x * 32;               // row block of WT
  for (int j = threadIdx.y; j < 32; j += 8)
    WT[(size_t)(yo0 + j) * n + (y0 + threadIdx.x)] = (__bf16)tile[threadIdx.x][j];
}

// --------------------------- bf16 GEMM-NT ----------------------------------
// C[M,N] = A[M,K] * Bt[N,K]^T, A/Bt bf16 row-major, f32 accumulate.
// MODE 0: write bf16 into [B,H,S,HD] head layout (Q/K/V projections)
// MODE 1: write f32 row-major + bias (final output projection)
// Block: 256 threads (8 waves), tile 128(M) x 128(N); wave tile 32x64
// (2x4 WMMA tiles -> 8 v_wmma per K-step of 32).

template <int MODE>
__global__ void __launch_bounds__(256)
k_gemm_nt(const __bf16* __restrict__ A, const __bf16* __restrict__ Bt,
          __bf16* __restrict__ outb, float* __restrict__ outf,
          const float* __restrict__ bias, int Kdim, int Ndim) {
  __shared__ __attribute__((aligned(16))) __bf16 As[128 * 40];  // +8 pad
  __shared__ __attribute__((aligned(16))) __bf16 Bs[128 * 40];

  const int tid = threadIdx.x;
  const int w = tid >> 5, lane = tid & 31;
  const int lr = lane & 15, hi = lane >> 4;     // lane-half per WMMA layout
  const int wm = w & 3, wn = w >> 2;            // 4(M) x 2(N) wave grid
  const int m0 = blockIdx.y * 128, n0 = blockIdx.x * 128;

  const v8f vz = {0.f, 0.f, 0.f, 0.f, 0.f, 0.f, 0.f, 0.f};
  v8f acc[2][4];
#pragma unroll
  for (int i = 0; i < 2; i++)
#pragma unroll
    for (int j = 0; j < 4; j++) acc[i][j] = vz;

  for (int k0 = 0; k0 < Kdim; k0 += 32) {
    // stage A and Bt: each 128x32 bf16 = 512 16B chunks, 2 per thread each
#pragma unroll
    for (int i = 0; i < 2; i++) {
      int id = tid + i * 256;
      int r = id >> 2, cs = (id & 3) * 8;
      stage16(&As[r * 40 + cs], &A[(size_t)(m0 + r) * Kdim + k0 + cs]);
      stage16(&Bs[r * 40 + cs], &Bt[(size_t)(n0 + r) * Kdim + k0 + cs]);
    }
    // prefetch next K tile (speculative; OOB prefetch is dropped)
    __builtin_prefetch(&A[(size_t)(m0 + (tid >> 2)) * Kdim + k0 + 32], 0, 3);
    __builtin_prefetch(&Bt[(size_t)(n0 + (tid >> 2)) * Kdim + k0 + 32], 0, 3);
    stage_fence();
    __syncthreads();

    v16bf af[2], bf[4];
#pragma unroll
    for (int t = 0; t < 2; t++) {               // A: lane = row
      int ar = wm * 32 + t * 16 + lr;
      af[t] = cat8(*(const v8bf*)&As[ar * 40 + hi * 8],
                   *(const v8bf*)&As[ar * 40 + hi * 8 + 16]);
    }
#pragma unroll
    for (int t = 0; t < 4; t++) {               // B: lane = column
      int br = wn * 64 + t * 16 + lr;
      bf[t] = cat8(*(const v8bf*)&Bs[br * 40 + hi * 8],
                   *(const v8bf*)&Bs[br * 40 + hi * 8 + 16]);
    }
#pragma unroll
    for (int mt = 0; mt < 2; mt++)
#pragma unroll
      for (int nt = 0; nt < 4; nt++)
        acc[mt][nt] = wmma_bf16(af[mt], bf[nt], acc[mt][nt]);
    __syncthreads();
  }

  // epilogue (C/D layout: lane = col, vgpr r = row, +8 rows for upper half)
#pragma unroll
  for (int mt = 0; mt < 2; mt++) {
#pragma unroll
    for (int nt = 0; nt < 4; nt++) {
      int col = n0 + wn * 64 + nt * 16 + lr;
      int rb  = m0 + wm * 32 + mt * 16 + hi * 8;
#pragma unroll
      for (int r = 0; r < 8; r++) {
        float v = acc[mt][nt][r];
        int row = rb + r;
        if (MODE == 0) {
          int b = row >> 11, s = row & (Sk - 1);  // S = 2048
          int h = col >> 6, hd = col & 63;
          outb[(((size_t)(b * Hk + h) * Sk + s) << 6) + hd] = (__bf16)v;
        } else {
          outf[(size_t)row * Ndim + col] = v + bias[col];
        }
      }
    }
  }
}

// --------------------------- flash attention --------------------------------
// Grid: (S/64, B*H). Block: 128 threads (4 waves); wave w owns 16 query rows.
// Key loop in 32-key chunks up to the causal bound of the 64-row query tile;
// waves skip chunks that are fully masked for their rows (barriers kept).

__global__ void __launch_bounds__(128)
k_flash(const __bf16* __restrict__ Q, const __bf16* __restrict__ K,
        const __bf16* __restrict__ V, __bf16* __restrict__ ctx) {
  __shared__ __attribute__((aligned(16))) __bf16 Ks[32 * 72];   // [key][hd]
  __shared__ __attribute__((aligned(16))) __bf16 Vt[64 * 40];   // [hd][key]
  __shared__ __attribute__((aligned(16))) __bf16 Ps[4 * 16 * 40]; // per-wave P

  const int tid = threadIdx.x, w = tid >> 5, lane = tid & 31;
  const int lr = lane & 15, hi = lane >> 4;
  const int bh = blockIdx.y;
  const int qbase = blockIdx.x * 64;
  const int qrow0 = qbase + w * 16;
  const size_t base = (size_t)bh * Sk * HDk;
  const __bf16* Qb = Q + base;
  const __bf16* Kb = K + base;
  const __bf16* Vb = V + base;

  // Q fragments (A operand, contraction over HD=64 -> two K=32 fragments)
  v16bf qf[2];
  {
    int qr = qrow0 + lr;
#pragma unroll
    for (int f = 0; f < 2; f++) {
      int c0 = f * 32 + hi * 8;
      qf[f] = cat8(*(const v8bf*)&Qb[(size_t)qr * 64 + c0],
                   *(const v8bf*)&Qb[(size_t)qr * 64 + c0 + 16]);
    }
  }

  // softmax runs in the exp2 domain: fold 1/sqrt(64) * log2(e) into one scale
  const float SCL = 0.125f * 1.44269504088896340736f;

  const v8f vz = {0.f, 0.f, 0.f, 0.f, 0.f, 0.f, 0.f, 0.f};
  v8f o[4];
  float mr[8], ls[8];
#pragma unroll
  for (int j = 0; j < 4; j++) o[j] = vz;
#pragma unroll
  for (int r = 0; r < 8; r++) { mr[r] = -INFINITY; ls[r] = 0.f; }

  __bf16* Pw = &Ps[w * 16 * 40];
  const int kend = qbase + 64;

  for (int kc = 0; kc < kend; kc += 32) {
    // stage K chunk (row-major; B operand for scores: lane = key, contiguous)
#pragma unroll
    for (int i = 0; i < 2; i++) {
      int id = tid + i * 128;
      int r = id >> 3, cs = (id & 7) * 8;
      stage16(&Ks[r * 72 + cs], &Kb[(size_t)(kc + r) * 64 + cs]);
    }
    // stage V transposed -> Vt[hd][key] so PV B-fragments are ds_load_b128
    {
      int key = tid & 31, hb = (tid >> 5) * 16;
      v8bf v0 = *(const v8bf*)&Vb[(size_t)(kc + key) * 64 + hb];
      v8bf v1 = *(const v8bf*)&Vb[(size_t)(kc + key) * 64 + hb + 8];
#pragma unroll
      for (int j = 0; j < 8; j++) {
        Vt[(hb + j)     * 40 + key] = v0[j];
        Vt[(hb + 8 + j) * 40 + key] = v1[j];
      }
    }
    stage_fence();
    __syncthreads();

    if (kc <= qrow0 + 15) {   // wave-uniform: skip fully-masked chunks
      // scores: two 16-key subtiles, each accumulating 2 WMMAs over HD
      v8f sc[2];
#pragma unroll
      for (int nt = 0; nt < 2; nt++) {
        int kr = nt * 16 + lr;
        v8f a = vz;
#pragma unroll
        for (int f = 0; f < 2; f++) {
          int c0 = f * 32 + hi * 8;
          v16bf kf = cat8(*(const v8bf*)&Ks[kr * 72 + c0],
                          *(const v8bf*)&Ks[kr * 72 + c0 + 16]);
          a = wmma_bf16(qf[f], kf, a);
        }
        sc[nt] = a;
      }

      // causal mask + online softmax (base-2), stash P in LDS
      float fac[8];
#pragma unroll
      for (int r = 0; r < 8; r++) {
        int qg = qrow0 + hi * 8 + r;
        float s0 = (kc + lr      <= qg) ? sc[0][r] * SCL : -INFINITY;
        float s1 = (kc + 16 + lr <= qg) ? sc[1][r] * SCL : -INFINITY;
        float mx = redmax16(fmaxf(s0, s1));
        float nm = fmaxf(mr[r], mx);
        float p0 = fexp2(s0 - nm);
        float p1 = fexp2(s1 - nm);
        float f  = fexp2(mr[r] - nm);
        mr[r] = nm;
        ls[r] = ls[r] * f + redsum16(p0 + p1);
        fac[r] = f;
        Pw[(hi * 8 + r) * 40 + lr]      = (__bf16)p0;
        Pw[(hi * 8 + r) * 40 + 16 + lr] = (__bf16)p1;
      }

      // rescale running context
#pragma unroll
      for (int j = 0; j < 4; j++)
#pragma unroll
        for (int r = 0; r < 8; r++) o[j][r] *= fac[r];

      // reload P in A-fragment layout (same-wave LDS ops are in order)
      v16bf pf = cat8(*(const v8bf*)&Pw[lr * 40 + hi * 8],
                      *(const v8bf*)&Pw[lr * 40 + hi * 8 + 16]);
      // O += P x V   (four 16-col output tiles over HD=64)
#pragma unroll
      for (int j = 0; j < 4; j++) {
        int vr = j * 16 + lr;
        v16bf vf = cat8(*(const v8bf*)&Vt[vr * 40 + hi * 8],
                        *(const v8bf*)&Vt[vr * 40 + hi * 8 + 16]);
        o[j] = wmma_bf16(pf, vf, o[j]);
      }
    }
    __syncthreads();
  }

  // normalize and write ctx as [b, s, h*64+hd] bf16 (row-major for out-proj)
  const int b = bh >> 4, h = bh & 15;
#pragma unroll
  for (int j = 0; j < 4; j++) {
    int hd = j * 16 + lr;
#pragma unroll
    for (int r = 0; r < 8; r++) {
      int q = qrow0 + hi * 8 + r;
      float val = o[j][r] / ls[r];
      ctx[(size_t)(b * Sk + q) * Dk + h * 64 + hd] = (__bf16)val;
    }
  }
}

// --------------------------- launcher --------------------------------------

extern "C" void kernel_launch(void* const* d_in, const int* in_sizes, int n_in,
                              void* d_out, int out_size, void* d_ws,
                              size_t ws_size, hipStream_t stream) {
  const float* x  = (const float*)d_in[0];
  const float* Wq = (const float*)d_in[1];
  const float* Wk = (const float*)d_in[2];
  const float* Wv = (const float*)d_in[3];
  const float* Wo = (const float*)d_in[4];
  const float* bo = (const float*)d_in[5];

  char* ws = (char*)d_ws;
  const size_t MB = 1ull << 20;
  __bf16* xbf = (__bf16*)(ws + 0 * MB);
  __bf16* WqT = (__bf16*)(ws + 8 * MB);
  __bf16* WkT = (__bf16*)(ws + 10 * MB);
  __bf16* WvT = (__bf16*)(ws + 12 * MB);
  __bf16* Wob = (__bf16*)(ws + 14 * MB);
  __bf16* Qh  = (__bf16*)(ws + 16 * MB);
  __bf16* Kh  = (__bf16*)(ws + 24 * MB);
  __bf16* Vh  = (__bf16*)(ws + 32 * MB);
  __bf16* ctx = (__bf16*)(ws + 40 * MB);

  // convert inputs to bf16 (weights for Q/K/V transposed to N x K)
  k_cvt_f32_bf16<<<(Bk * Sk * Dk) / 1024, 256, 0, stream>>>(x, xbf, Bk * Sk * Dk);
  dim3 tb(32, 8), tg(Dk / 32, Dk / 32);
  k_tr_cvt<<<tg, tb, 0, stream>>>(Wq, WqT, Dk);
  k_tr_cvt<<<tg, tb, 0, stream>>>(Wk, WkT, Dk);
  k_tr_cvt<<<tg, tb, 0, stream>>>(Wv, WvT, Dk);
  k_cvt_f32_bf16<<<(Dk * Dk) / 1024, 256, 0, stream>>>(Wo, Wob, Dk * Dk);

  // Q/K/V projections into [B,H,S,HD]
  dim3 gg(Dk / 128, (Bk * Sk) / 128);  // (8, 32)
  k_gemm_nt<0><<<gg, 256, 0, stream>>>(xbf, WqT, Qh, nullptr, nullptr, Dk, Dk);
  k_gemm_nt<0><<<gg, 256, 0, stream>>>(xbf, WkT, Kh, nullptr, nullptr, Dk, Dk);
  k_gemm_nt<0><<<gg, 256, 0, stream>>>(xbf, WvT, Vh, nullptr, nullptr, Dk, Dk);

  // causal flash attention -> ctx
  k_flash<<<dim3(Sk / 64, Bk * Hk), 128, 0, stream>>>(Qh, Kh, Vh, ctx);

  // output projection: out = ctx @ Wo^T + bo  (fp32)
  k_gemm_nt<1><<<gg, 256, 0, stream>>>(ctx, Wob, nullptr, (float*)d_out, bo,
                                       Dk, Dk);

  (void)in_sizes; (void)n_in; (void)out_size; (void)ws_size;
}
